// SparseGroupingLayer_29111288332439
// MI455X (gfx1250) — matrix operand order repaired and checked by reading, not compiled, not run on hardware
//
#include <hip/hip_runtime.h>
#include <hip/hip_bf16.h>
#include <math.h>

// ---------------------------------------------------------------------------
// SparseGroupingLayer for MI455X (gfx1250), wave32 + WMMA bf16.
// All 128-deep matmuls (k/v/q projections, 3x3 stride-2 conv, MLP) go through
// v_wmma_f32_16x16x32_bf16.  Tile fills use GLOBAL_LOAD_ASYNC_TO_LDS_B128
// (ASYNCcnt) when the toolchain exposes the builtin, else global->VGPR->LDS.
// LN / softmax / segment-sum are f32 VALU kernels.
// ---------------------------------------------------------------------------

#define DIMC   128
#define BATCH  4
#define HOC    64
#define WOC    64
#define NPOS   (HOC * WOC)     // 4096 output positions per batch
#define NPIX   (128 * 128)     // 16384 input pixels per batch
#define MT     (BATCH * NPIX)  // 65536 total input rows
#define MO     (BATCH * NPOS)  // 16384 total output rows

typedef __attribute__((ext_vector_type(16))) __bf16 v16bf;
typedef __attribute__((ext_vector_type(8)))  float  v8f;

union AFrag { v16bf v; unsigned int u[8]; };
union CFrag { v8f   v; float        f[8]; };

#ifndef __has_builtin
#define __has_builtin(x) 0
#endif

#if __has_builtin(__builtin_amdgcn_global_load_async_to_lds_b128)
#define HAVE_ASYNC_LDS 1
typedef __attribute__((ext_vector_type(4))) int v4i_async;
typedef __attribute__((address_space(1))) v4i_async as1_v4i;
typedef __attribute__((address_space(3))) v4i_async as3_v4i;
#else
#define HAVE_ASYNC_LDS 0
#endif

__device__ __forceinline__ void wait_async_lds() {
#if HAVE_ASYNC_LDS
#if __has_builtin(__builtin_amdgcn_s_wait_asynccnt)
  __builtin_amdgcn_s_wait_asynccnt(0);
#else
  asm volatile("s_wait_asynccnt 0" ::: "memory");
#endif
#endif
}

__device__ __forceinline__ float wave_sum(float v) {
  #pragma unroll
  for (int off = 16; off >= 1; off >>= 1)
    v += __shfl_xor(v, off, 32);
  return v;
}

// ---------------------------------------------------------------------------
// Generic bf16 GEMM:  C[M,N] = A[M,K] @ W[K,N]   (+bias, +exact GELU)
// block = 256 threads (8 waves); each block does a 128x128 C tile;
// wave w owns rows [16w,16w+16) of the tile across all 8 N-subtiles.
// mode: 0 = plain, 1 = +bias, 2 = gelu(x + bias)
// ---------------------------------------------------------------------------
__global__ __launch_bounds__(256)
void gemm_bf16_wmma(const __bf16* __restrict__ A, const __bf16* __restrict__ W,
                    const float* __restrict__ bias, float* __restrict__ C,
                    int M, int K, int N, int mode)
{
  __shared__ __align__(16) __bf16 As[128 * 32];
  __shared__ __align__(16) __bf16 Bs[32 * 128];
  const int tid  = threadIdx.x;
  const int lane = tid & 31;
  const int wv   = tid >> 5;
  const int m0   = blockIdx.y * 128;
  const int n0   = blockIdx.x * 128;

  v8f zero = {0.f, 0.f, 0.f, 0.f, 0.f, 0.f, 0.f, 0.f};
  v8f acc[8];
  #pragma unroll
  for (int i = 0; i < 8; ++i) acc[i] = zero;

  const int ksteps = K >> 5;
  for (int kt = 0; kt < ksteps; ++kt) {
#if HAVE_ASYNC_LDS
    // A tile 128x32 and B tile 32x128 streamed straight into LDS (ASYNCcnt)
    #pragma unroll
    for (int i = 0; i < 2; ++i) {
      int e = (i * 256 + tid) * 8;
      int r = e >> 5, c = e & 31;
      __builtin_amdgcn_global_load_async_to_lds_b128(
          (as1_v4i*)&A[(size_t)(m0 + r) * K + kt * 32 + c],
          (as3_v4i*)&As[e], 0, 0);
    }
    #pragma unroll
    for (int i = 0; i < 2; ++i) {
      int e = (i * 256 + tid) * 8;
      int r = e >> 7, c = e & 127;
      __builtin_amdgcn_global_load_async_to_lds_b128(
          (as1_v4i*)&W[(size_t)(kt * 32 + r) * N + n0 + c],
          (as3_v4i*)&Bs[e], 0, 0);
    }
    wait_async_lds();
#else
    // A tile 128x32 (row-major [m][k])
    #pragma unroll
    for (int i = 0; i < 4; ++i) {
      int e = (i * 256 + tid) * 4;
      int r = e >> 5, c = e & 31;
      *(uint2*)&As[e] =
          *(const uint2*)&A[(size_t)(m0 + r) * K + kt * 32 + c];
    }
    // B tile 32x128 (row-major [k][n])
    #pragma unroll
    for (int i = 0; i < 4; ++i) {
      int e = (i * 256 + tid) * 4;
      int r = e >> 7, c = e & 127;
      *(uint2*)&Bs[e] =
          *(const uint2*)&W[(size_t)(kt * 32 + r) * N + n0 + c];
    }
#endif
    __syncthreads();

    // A fragment per ISA 16-bit A 16x32 layout
    AFrag af;
    {
      const unsigned int* As32 =
          (const unsigned int*)&As[(wv * 16 + (lane & 15)) * 32];
      const int kbase = (lane >> 4) * 8;
      #pragma unroll
      for (int r = 0; r < 8; ++r) {
        int k = ((r & 4) ? 16 : 0) + kbase + 2 * (r & 3);
        af.u[r] = As32[k >> 1];
      }
    }
    #pragma unroll
    for (int nt = 0; nt < 8; ++nt) {
      AFrag bfr;
      const int kb = (lane >> 4) * 8;
      #pragma unroll
      for (int s = 0; s < 16; ++s)
        bfr.v[s] = Bs[(((s & 8) ? 16 : 0) + kb + (s & 7)) * 128 + nt * 16 + (lane & 15)];
      acc[nt] = __builtin_amdgcn_wmma_f32_16x16x32_bf16(
          false, af.v, false, bfr.v, (short)0, acc[nt], false, false);
    }
    __syncthreads();
  }

  // C layout: VGPR r -> M = r (lanes 0-15) / r+8 (lanes 16-31), N = lane&15
  const int rowoff = (lane >> 4) * 8;
  const int col    = lane & 15;
  #pragma unroll
  for (int nt = 0; nt < 8; ++nt) {
    CFrag cf; cf.v = acc[nt];
    int n = n0 + nt * 16 + col;
    float bv = bias ? bias[n] : 0.f;
    #pragma unroll
    for (int r = 0; r < 8; ++r) {
      int m = m0 + wv * 16 + rowoff + r;
      float val = cf.f[r];
      if (mode >= 1) val += bv;
      if (mode == 2) val = 0.5f * val * (1.0f + erff(val * 0.70710678118f));
      C[(size_t)m * N + n] = val;
    }
  }
}

// ---------------------------------------------------------------------------
// Stride-2 3x3 conv as 9 gathered GEMMs accumulated in WMMA registers.
// grid.x = BATCH * 32; each block: 128 output positions x 128 channels.
// Border taps are zero-padded: valid rows stream via async-to-LDS, invalid
// rows are zero-filled with ds stores.
// ---------------------------------------------------------------------------
__global__ __launch_bounds__(256)
void conv_seed_wmma(const __bf16* __restrict__ Xb, const __bf16* __restrict__ CW,
                    float* __restrict__ seed)
{
  __shared__ __align__(16) __bf16 As[128 * 32];
  __shared__ __align__(16) __bf16 Bs[32 * 128];
  const int tid  = threadIdx.x;
  const int lane = tid & 31;
  const int wv   = tid >> 5;
  const int b    = blockIdx.x >> 5;
  const int p0   = (blockIdx.x & 31) * 128;

  v8f zero = {0.f, 0.f, 0.f, 0.f, 0.f, 0.f, 0.f, 0.f};
  v8f acc[8];
  #pragma unroll
  for (int i = 0; i < 8; ++i) acc[i] = zero;

  for (int tap = 0; tap < 9; ++tap) {
    const int dy = tap / 3 - 1, dx = tap % 3 - 1;
    for (int kt = 0; kt < 4; ++kt) {
#if HAVE_ASYNC_LDS
      #pragma unroll
      for (int i = 0; i < 2; ++i) {
        int e = (i * 256 + tid) * 8;
        int r = e >> 5, c = e & 31;
        int p  = p0 + r;
        int ho = p >> 6, wo = p & 63;
        int ii = 2 * ho + dy, jj = 2 * wo + dx;
        if (ii >= 0 && ii < 128 && jj >= 0 && jj < 128) {
          __builtin_amdgcn_global_load_async_to_lds_b128(
              (as1_v4i*)&Xb[((size_t)b * NPIX + ii * 128 + jj) * DIMC +
                            kt * 32 + c],
              (as3_v4i*)&As[e], 0, 0);
        } else {
          *(uint4*)&As[e] = make_uint4(0u, 0u, 0u, 0u);
        }
      }
      #pragma unroll
      for (int i = 0; i < 2; ++i) {
        int e = (i * 256 + tid) * 8;
        int r = e >> 7, c = e & 127;
        __builtin_amdgcn_global_load_async_to_lds_b128(
            (as1_v4i*)&CW[(size_t)tap * DIMC * DIMC + (kt * 32 + r) * DIMC + c],
            (as3_v4i*)&Bs[e], 0, 0);
      }
      wait_async_lds();
#else
      #pragma unroll
      for (int i = 0; i < 4; ++i) {
        int e = (i * 256 + tid) * 4;
        int r = e >> 5, c = e & 31;
        int p  = p0 + r;
        int ho = p >> 6, wo = p & 63;
        int ii = 2 * ho + dy, jj = 2 * wo + dx;
        uint2 val = make_uint2(0u, 0u);
        if (ii >= 0 && ii < 128 && jj >= 0 && jj < 128)
          val = *(const uint2*)
              &Xb[((size_t)b * NPIX + ii * 128 + jj) * DIMC + kt * 32 + c];
        *(uint2*)&As[e] = val;
      }
      #pragma unroll
      for (int i = 0; i < 4; ++i) {
        int e = (i * 256 + tid) * 4;
        int r = e >> 7, c = e & 127;
        *(uint2*)&Bs[e] = *(const uint2*)
            &CW[(size_t)tap * DIMC * DIMC + (kt * 32 + r) * DIMC + c];
      }
#endif
      __syncthreads();

      AFrag af;
      {
        const unsigned int* As32 =
            (const unsigned int*)&As[(wv * 16 + (lane & 15)) * 32];
        const int kbase = (lane >> 4) * 8;
        #pragma unroll
        for (int r = 0; r < 8; ++r) {
          int k = ((r & 4) ? 16 : 0) + kbase + 2 * (r & 3);
          af.u[r] = As32[k >> 1];
        }
      }
      #pragma unroll
      for (int nt = 0; nt < 8; ++nt) {
        AFrag bfr;
        const int kb = (lane >> 4) * 8;
        #pragma unroll
        for (int s = 0; s < 16; ++s)
          bfr.v[s] = Bs[(((s & 8) ? 16 : 0) + kb + (s & 7)) * 128 + nt * 16 + (lane & 15)];
        acc[nt] = __builtin_amdgcn_wmma_f32_16x16x32_bf16(
            false, af.v, false, bfr.v, (short)0, acc[nt], false, false);
      }
      __syncthreads();
    }
  }

  const int rowoff = (lane >> 4) * 8;
  const int col    = lane & 15;
  #pragma unroll
  for (int nt = 0; nt < 8; ++nt) {
    CFrag cf; cf.v = acc[nt];
    int n = nt * 16 + col;
    #pragma unroll
    for (int r = 0; r < 8; ++r) {
      int p = p0 + wv * 16 + rowoff + r;
      seed[((size_t)b * NPOS + p) * DIMC + n] = cf.f[r];
    }
  }
}

// ---------------------------------------------------------------------------
// LayerNorm over DIM=128, one wave per row (4 elems/lane).
// outf: optional f32 output (addmode=1 -> accumulate), outb: optional bf16
// LN output, outraw: optional bf16 cast of the *un-normalized* input.
// ---------------------------------------------------------------------------
__global__ __launch_bounds__(256)
void ln_kernel(const float* __restrict__ in, const float* __restrict__ gg,
               const float* __restrict__ bb, float* __restrict__ outf,
               __bf16* __restrict__ outb, __bf16* __restrict__ outraw,
               int rows, int addmode)
{
  const int row  = blockIdx.x * 8 + (threadIdx.x >> 5);
  const int lane = threadIdx.x & 31;
  if (row >= rows) return;
  const size_t base = (size_t)row * DIMC + lane * 4;
  float v[4];
  #pragma unroll
  for (int c = 0; c < 4; ++c) v[c] = in[base + c];
  if (outraw) {
    #pragma unroll
    for (int c = 0; c < 4; ++c) outraw[base + c] = (__bf16)v[c];
  }
  float s = v[0] + v[1] + v[2] + v[3];
  s = wave_sum(s);
  const float mu = s * (1.0f / DIMC);
  float q = 0.f;
  #pragma unroll
  for (int c = 0; c < 4; ++c) { float d = v[c] - mu; q += d * d; }
  q = wave_sum(q);
  const float inv = rsqrtf(q * (1.0f / DIMC) + 1e-5f);
  #pragma unroll
  for (int c = 0; c < 4; ++c) {
    float y = (v[c] - mu) * inv * gg[lane * 4 + c] + bb[lane * 4 + c];
    if (outf) {
      if (addmode) outf[base + c] += y;
      else         outf[base + c]  = y;
    }
    if (outb) outb[base + c] = (__bf16)y;
  }
}

// ---------------------------------------------------------------------------
// Small elementwise helpers
// ---------------------------------------------------------------------------
__global__ void cast_f32_bf16(const float* __restrict__ in,
                              __bf16* __restrict__ out, int n)
{
  int i = blockIdx.x * 256 + threadIdx.x;
  if (i < n) out[i] = (__bf16)in[i];
}

__global__ void zero_f32(float* __restrict__ p, int n)
{
  int i = blockIdx.x * 256 + threadIdx.x;
  if (i < n) p[i] = 0.f;
}

// conv_w (O,I,3,3) -> bf16 [tap][i][o]
__global__ void conv_w_transpose(const float* __restrict__ cw,
                                 __bf16* __restrict__ out, int n)
{
  int idx = blockIdx.x * 256 + threadIdx.x;
  if (idx >= n) return;
  int tap = idx / (DIMC * DIMC);
  int rem = idx - tap * DIMC * DIMC;
  int i = rem / DIMC;
  int o = rem - i * DIMC;
  out[idx] = (__bf16)cw[(size_t)o * DIMC * 9 + i * 9 + tap];
}

// ---------------------------------------------------------------------------
// Attention logits + softmax (+eps), with atomic segment-sum into col_sums.
// One wave per (b, g, position).  65536 items total.
// ---------------------------------------------------------------------------
__global__ __launch_bounds__(256)
void attn_kernel(const float* __restrict__ kbuf, const float* __restrict__ qbuf,
                 const float* __restrict__ rpb, const float* __restrict__ tau,
                 float* __restrict__ attn_out, float* __restrict__ colsum)
{
  const int item = blockIdx.x * 8 + (threadIdx.x >> 5);
  const int lane = threadIdx.x & 31;
  const int b  = item >> 14;
  const int g  = (item >> 12) & 3;
  const int hw = item & 4095;
  const int h = hw >> 6, w = hw & 63;
  const int gy = g >> 1, gx = g & 1;

  const float* kv = kbuf +
      ((size_t)b * NPIX + (2 * h + gy) * 128 + (2 * w + gx)) * DIMC + lane * 4;
  const float k0 = kv[0], k1 = kv[1], k2 = kv[2], k3 = kv[3];

  const int ci = min(max(h, 1), 62);
  const int cj = min(max(w, 1), 62);
  const float scale = expf(tau[0]);

  float lg[9];
  int qpos[9];
  #pragma unroll
  for (int t = 0; t < 9; ++t) {
    int nh = ci + t / 3 - 1, nw = cj + t % 3 - 1;
    qpos[t] = nh * 64 + nw;
    const float* qv = qbuf + ((size_t)b * NPOS + qpos[t]) * DIMC + lane * 4;
    float d = k0 * qv[0] + k1 * qv[1] + k2 * qv[2] + k3 * qv[3];
    d = wave_sum(d);
    lg[t] = (d + rpb[g * 9 + t]) * scale;
  }
  float m = lg[0];
  #pragma unroll
  for (int t = 1; t < 9; ++t) m = fmaxf(m, lg[t]);
  float se = 0.f, ev[9];
  #pragma unroll
  for (int t = 0; t < 9; ++t) { ev[t] = expf(lg[t] - m); se += ev[t]; }
  const float inv = 1.0f / se;

  if (lane < 9) {
    float a = 0.f;
    int qp = 0;
    #pragma unroll
    for (int t = 0; t < 9; ++t)
      if (t == lane) { a = ev[t] * inv + 1e-6f; qp = qpos[t]; }
    attn_out[((size_t)(b * 4 + g) * NPOS + hw) * 9 + lane] = a;
    atomicAdd(&colsum[b * NPOS + qp], a);
  }
}

// ---------------------------------------------------------------------------
// A_col = attn / (col_sums[q_idx] + 1e-8); upd = sum_{g,t} A_col * v_nb;
// x_out += upd.  One wave per (b, position).  16384 items total.
// ---------------------------------------------------------------------------
__global__ __launch_bounds__(256)
void acol_upd_kernel(const float* __restrict__ attn_in,
                     const float* __restrict__ colsum,
                     const float* __restrict__ vbuf,
                     float* __restrict__ acol_out,
                     float* __restrict__ xout)
{
  const int item = blockIdx.x * 8 + (threadIdx.x >> 5);
  const int lane = threadIdx.x & 31;
  const int b  = item >> 12;
  const int hw = item & 4095;
  const int h = hw >> 6, w = hw & 63;
  const int ci = min(max(h, 1), 62);
  const int cj = min(max(w, 1), 62);

  float den[9];
  int nh[9], nw[9];
  #pragma unroll
  for (int t = 0; t < 9; ++t) {
    nh[t] = ci + t / 3 - 1;
    nw[t] = cj + t % 3 - 1;
    den[t] = colsum[b * NPOS + nh[t] * 64 + nw[t]] + 1e-8f;
  }

  float a0 = 0.f, a1 = 0.f, a2 = 0.f, a3 = 0.f;
  for (int g = 0; g < 4; ++g) {
    const int gy = g >> 1, gx = g & 1;
    #pragma unroll
    for (int t = 0; t < 9; ++t) {
      size_t ai = ((size_t)(b * 4 + g) * NPOS + hw) * 9 + t;
      float a  = attn_in[ai];
      float ac = a / den[t];
      if (lane == 0) acol_out[ai] = ac;
      const float* vv = vbuf +
          ((size_t)b * NPIX + (2 * nh[t] + gy) * 128 + (2 * nw[t] + gx)) * DIMC
          + lane * 4;
      a0 += ac * vv[0]; a1 += ac * vv[1]; a2 += ac * vv[2]; a3 += ac * vv[3];
    }
  }
  float* xp = xout + ((size_t)b * NPOS + hw) * DIMC + lane * 4;
  xp[0] += a0; xp[1] += a1; xp[2] += a2; xp[3] += a3;
}

// ---------------------------------------------------------------------------
// Host-side orchestration
// ---------------------------------------------------------------------------
extern "C" void kernel_launch(void* const* d_in, const int* in_sizes, int n_in,
                              void* d_out, int out_size, void* d_ws, size_t ws_size,
                              hipStream_t stream)
{
  (void)in_sizes; (void)n_in; (void)out_size; (void)ws_size;
  const float* x      = (const float*)d_in[0];
  const float* conv_w = (const float*)d_in[1];
  const float* q_w    = (const float*)d_in[2];
  const float* k_w    = (const float*)d_in[3];
  const float* v_w    = (const float*)d_in[4];
  const float* w1     = (const float*)d_in[5];
  const float* b1     = (const float*)d_in[6];
  const float* w2     = (const float*)d_in[7];
  const float* b2     = (const float*)d_in[8];
  const float* lng_i  = (const float*)d_in[9];
  const float* lnb_i  = (const float*)d_in[10];
  const float* lng_o  = (const float*)d_in[11];
  const float* lnb_o  = (const float*)d_in[12];
  const float* tau    = (const float*)d_in[13];
  const float* rpb    = (const float*)d_in[14];

  float* xout     = (float*)d_out;                          // (B,4096,128) f32
  float* attn_out = xout + (size_t)MO * DIMC;               // (B,4,4096,9)
  float* acol_out = attn_out + (size_t)BATCH * 4 * NPOS * 9;

  char* ws = (char*)d_ws;
  size_t off = 0;
  auto alloc = [&](size_t bytes) -> void* {
    void* p = ws + off;
    off = (off + bytes + 255) & ~(size_t)255;
    return p;
  };

  __bf16* xn_bf  = (__bf16*)alloc((size_t)MT * DIMC * 2);   // LN_in(x) bf16
  __bf16* xr_bf  = (__bf16*)alloc((size_t)MT * DIMC * 2);   // raw x bf16
  float*  kbuf   = (float*) alloc((size_t)MT * DIMC * 4);
  float*  vbuf   = (float*) alloc((size_t)MT * DIMC * 4);
  float*  seed   = (float*) alloc((size_t)MO * DIMC * 4);
  float*  qbuf   = (float*) alloc((size_t)MO * DIMC * 4);
  __bf16* xo_bf  = (__bf16*)alloc((size_t)MO * DIMC * 2);
  float*  h1     = (float*) alloc((size_t)MO * 2 * DIMC * 4);
  __bf16* h1_bf  = (__bf16*)alloc((size_t)MO * 2 * DIMC * 2);
  float*  h2     = (float*) alloc((size_t)MO * DIMC * 4);
  float*  colsum = (float*) alloc((size_t)MO * 4);
  __bf16* kw_bf  = (__bf16*)alloc((size_t)DIMC * DIMC * 2);
  __bf16* qw_bf  = (__bf16*)alloc((size_t)DIMC * DIMC * 2);
  __bf16* vw_bf  = (__bf16*)alloc((size_t)DIMC * DIMC * 2);
  __bf16* w1_bf  = (__bf16*)alloc((size_t)DIMC * 2 * DIMC * 2);
  __bf16* w2_bf  = (__bf16*)alloc((size_t)2 * DIMC * DIMC * 2);
  __bf16* cw_bf  = (__bf16*)alloc((size_t)9 * DIMC * DIMC * 2);

  // --- weight precision conversion ---
  cast_f32_bf16<<<(DIMC * DIMC + 255) / 256, 256, 0, stream>>>(k_w, kw_bf, DIMC * DIMC);
  cast_f32_bf16<<<(DIMC * DIMC + 255) / 256, 256, 0, stream>>>(q_w, qw_bf, DIMC * DIMC);
  cast_f32_bf16<<<(DIMC * DIMC + 255) / 256, 256, 0, stream>>>(v_w, vw_bf, DIMC * DIMC);
  cast_f32_bf16<<<(2 * DIMC * DIMC + 255) / 256, 256, 0, stream>>>(w1, w1_bf, 2 * DIMC * DIMC);
  cast_f32_bf16<<<(2 * DIMC * DIMC + 255) / 256, 256, 0, stream>>>(w2, w2_bf, 2 * DIMC * DIMC);
  conv_w_transpose<<<(9 * DIMC * DIMC + 255) / 256, 256, 0, stream>>>(conv_w, cw_bf, 9 * DIMC * DIMC);

  // --- LN_in(x) -> bf16, raw x -> bf16 ---
  ln_kernel<<<MT / 8, 256, 0, stream>>>(x, lng_i, lnb_i, nullptr, xn_bf, xr_bf, MT, 0);

  // --- k and v projections (65536 x 128 x 128 WMMA GEMMs) ---
  gemm_bf16_wmma<<<dim3(1, MT / 128), 256, 0, stream>>>(xn_bf, kw_bf, nullptr, kbuf, MT, DIMC, DIMC, 0);
  gemm_bf16_wmma<<<dim3(1, MT / 128), 256, 0, stream>>>(xr_bf, vw_bf, nullptr, vbuf, MT, DIMC, DIMC, 0);

  // --- stride-2 3x3 conv -> seed, then x_out = LN_out(seed) ---
  conv_seed_wmma<<<BATCH * 32, 256, 0, stream>>>(xr_bf, cw_bf, seed);
  ln_kernel<<<MO / 8, 256, 0, stream>>>(seed, lng_o, lnb_o, xout, nullptr, nullptr, MO, 0);

  for (int iter = 0; iter < 3; ++iter) {
    // q = LN_out(x_out) @ q_w
    ln_kernel<<<MO / 8, 256, 0, stream>>>(xout, lng_o, lnb_o, nullptr, xo_bf, nullptr, MO, 0);
    gemm_bf16_wmma<<<dim3(1, MO / 128), 256, 0, stream>>>(xo_bf, qw_bf, nullptr, qbuf, MO, DIMC, DIMC, 0);

    // attention + column sums
    zero_f32<<<MO / 256, 256, 0, stream>>>(colsum, MO);
    attn_kernel<<<(BATCH * 4 * NPOS) / 8, 256, 0, stream>>>(kbuf, qbuf, rpb, tau, attn_out, colsum);
    acol_upd_kernel<<<MO / 8, 256, 0, stream>>>(attn_out, colsum, vbuf, acol_out, xout);

    // MLP: h = gelu(x_out @ w1 + b1) @ w2 + b2 ; x_out += LN_out(h)
    cast_f32_bf16<<<(MO * DIMC) / 256, 256, 0, stream>>>(xout, xo_bf, MO * DIMC);
    gemm_bf16_wmma<<<dim3(2, MO / 128), 256, 0, stream>>>(xo_bf, w1_bf, b1, h1, MO, DIMC, 2 * DIMC, 2);
    cast_f32_bf16<<<(MO * 2 * DIMC) / 256, 256, 0, stream>>>(h1, h1_bf, MO * 2 * DIMC);
    gemm_bf16_wmma<<<dim3(1, MO / 128), 256, 0, stream>>>(h1_bf, w2_bf, b2, h2, MO, 2 * DIMC, DIMC, 1);
    ln_kernel<<<MO / 8, 256, 0, stream>>>(h2, lng_o, lnb_o, xout, nullptr, nullptr, MO, 1);
  }
}
